// MinGRU_22273700397170
// MI455X (gfx1250) — compile-verified, matrix operand order checked by reading, not run on hardware
//
#include <hip/hip_runtime.h>
#include <hip/hip_bf16.h>
#include <math.h>

typedef __attribute__((ext_vector_type(16))) _Float16 v16h;
typedef __attribute__((ext_vector_type(8)))  _Float16 v8h;
typedef __attribute__((ext_vector_type(8)))  float    v8f;
typedef int v4i __attribute__((__vector_size__(16)));

#define B_  8
#define T_  4096
#define D_  1024
#define H_  1024
#define NC  32            // scan chunks
#define TC  (T_ / NC)     // 128 steps per chunk
#define LDS_STRIDE 40     // 32 halves + 8 pad (80B rows -> conflict-free b128)
#define KSTAGES (D_ / 32)

// ---------- math helpers ----------
__device__ __forceinline__ float sp(float x) {            // softplus, stable
    return fmaxf(x, 0.0f) + log1pf(expf(-fabsf(x)));
}
__device__ __forceinline__ float log_g(float x) {
    return (x >= 0.0f) ? logf(x + 0.5f) : -sp(-x);
}
__device__ __forceinline__ float g_fun(float x) {
    return (x >= 0.0f) ? (x + 0.5f) : (1.0f / (1.0f + expf(-x)));
}
__device__ __forceinline__ float lae(float a, float b) {  // logaddexp
    float mx = fmaxf(a, b);
    if (mx == -__builtin_inff()) return mx;
    float mn = fminf(a, b);
    return mx + log1pf(expf(mn - mx));
}

// ---------- CDNA5 async global->LDS copy (16B per lane) ----------
__device__ __forceinline__ void async_cp16(const _Float16* g, _Float16* l) {
    __builtin_amdgcn_global_load_async_to_lds_b128(
        (__attribute__((address_space(1))) v4i*)g,
        (__attribute__((address_space(3))) v4i*)l,
        0, 0);
}
__device__ __forceinline__ void wait_async0() {
    asm volatile("s_wait_asynccnt 0x0" ::: "memory");
}

// ---------- WMMA helpers ----------
__device__ __forceinline__ v8f wmma_f16(v16h a, v16h b, v8f c) {
    return __builtin_amdgcn_wmma_f32_16x16x32_f16(false, a, false, b,
                                                  (short)0, c, false, false);
}

// 16x32 f16 fragment (A layout; B identical because B tiles are staged n-major).
// Lanes 0-15: row=lane, K 0-7 + 16-23; lanes 16-31: row=lane&15, K 8-15 + 24-31.
__device__ __forceinline__ v16h ld_frag(const _Float16* p0, int lane) {
    const int r  = lane & 15;
    const int kh = (lane & 16) ? 8 : 0;
    const _Float16* p = p0 + r * LDS_STRIDE + kh;
    v8h lo = *(const v8h*)(p);
    v8h hi = *(const v8h*)(p + 16);
    v16h f;
#pragma unroll
    for (int i = 0; i < 8; ++i) { f[i] = lo[i]; f[8 + i] = hi[i]; }
    return f;
}

// ---------- Kernel 0a: x (f32) -> xh (f16), 8 elements/thread ----------
__global__ __launch_bounds__(256)
void MinGRU_cvtx_kernel(const float* __restrict__ x, _Float16* __restrict__ xh) {
    size_t i = ((size_t)blockIdx.x * 256 + threadIdx.x) * 8;
    float4 a = *(const float4*)(x + i);
    float4 b = *(const float4*)(x + i + 4);
    v8h o;
    o[0] = (_Float16)a.x; o[1] = (_Float16)a.y; o[2] = (_Float16)a.z; o[3] = (_Float16)a.w;
    o[4] = (_Float16)b.x; o[5] = (_Float16)b.y; o[6] = (_Float16)b.z; o[7] = (_Float16)b.w;
    *(v8h*)(xh + i) = o;
}

// ---------- Kernel 0b: W [K,N] f32 -> WT [N,K] f16 (LDS-tiled transpose) ------
__global__ __launch_bounds__(256)
void MinGRU_cvtw_kernel(const float* __restrict__ Wz, const float* __restrict__ Wh,
                        _Float16* __restrict__ wzt, _Float16* __restrict__ wht) {
    const float* W  = blockIdx.z ? Wh : Wz;
    _Float16*   WT  = blockIdx.z ? wht : wzt;
    __shared__ _Float16 t[32][36];
    int r  = threadIdx.x >> 3;         // 0..31
    int c4 = (threadIdx.x & 7) * 4;    // 0,4,..,28
    int k0 = blockIdx.x * 32, n0 = blockIdx.y * 32;
    float4 f = *(const float4*)(W + (size_t)(k0 + r) * H_ + n0 + c4);
    t[r][c4 + 0] = (_Float16)f.x; t[r][c4 + 1] = (_Float16)f.y;
    t[r][c4 + 2] = (_Float16)f.z; t[r][c4 + 3] = (_Float16)f.w;
    __syncthreads();
    _Float16* d = WT + (size_t)(n0 + r) * D_ + k0 + c4;
    d[0] = t[c4 + 0][r]; d[1] = t[c4 + 1][r];
    d[2] = t[c4 + 2][r]; d[3] = t[c4 + 3][r];
}

// ---------- Kernel 1: fused dual GEMM, async-staged, double-buffered ----------
// k = x@Wz + bz ; p = x@Wh + bh
// cbuf[b,t,h]  = -softplus(k)              (log_coeffs)
// out[b,t+1,h] = -softplus(-k) + log_g(p)  (log_values, overwritten by scan3)
__global__ __launch_bounds__(256)
void MinGRU_gemm_kernel(const _Float16* __restrict__ xh,
                        const _Float16* __restrict__ wzt,
                        const _Float16* __restrict__ wht,
                        const float* __restrict__ bz, const float* __restrict__ bh,
                        float* __restrict__ cbuf, float* __restrict__ out) {
    __shared__ _Float16 sA [2][128 * LDS_STRIDE];  // 128 M-rows x 32 K
    __shared__ _Float16 sBz[2][ 64 * LDS_STRIDE];  // 64 N-rows x 32 K
    __shared__ _Float16 sBh[2][ 64 * LDS_STRIDE];

    const int tid   = threadIdx.x;
    const int lane  = tid & 31;
    const int wv    = tid >> 5;
    const int waveM = wv & 3;
    const int waveN = wv >> 2;
    const int mBase = blockIdx.y * 128;
    const int nBase = blockIdx.x * 64;

    // issue one stage of async copies (4 x b128 per thread)
    auto stage = [&](int buf, int kb) {
#pragma unroll
        for (int i = 0; i < 2; ++i) {                    // A: 512 granules
            int lin = tid + i * 256;
            int row = lin >> 2;
            int g   = (lin & 3) * 8;
            async_cp16(xh + (size_t)(mBase + row) * D_ + kb + g,
                       &sA[buf][row * LDS_STRIDE + g]);
        }
        {                                                // B: 256 granules each
            int row = tid >> 2;
            int g   = (tid & 3) * 8;
            async_cp16(wzt + (size_t)(nBase + row) * D_ + kb + g,
                       &sBz[buf][row * LDS_STRIDE + g]);
            async_cp16(wht + (size_t)(nBase + row) * D_ + kb + g,
                       &sBh[buf][row * LDS_STRIDE + g]);
        }
    };

    v8f ak[2][2], ap[2][2];
#pragma unroll
    for (int i = 0; i < 2; ++i)
#pragma unroll
        for (int j = 0; j < 2; ++j) { ak[i][j] = (v8f)(0.0f); ap[i][j] = (v8f)(0.0f); }

    stage(0, 0);
    for (int s = 0; s < KSTAGES; ++s) {
        const int cur = s & 1;
        wait_async0();                // this wave's fills for buf[cur] landed
        __syncthreads();              // everyone's fills landed
        if (s + 1 < KSTAGES) stage(1 - cur, (s + 1) * 32);  // overlap next fill

        v16h a0  = ld_frag(&sA [cur][(waveM * 32 +  0) * LDS_STRIDE], lane);
        v16h a1  = ld_frag(&sA [cur][(waveM * 32 + 16) * LDS_STRIDE], lane);
        v16h bz0 = ld_frag(&sBz[cur][(waveN * 32 +  0) * LDS_STRIDE], lane);
        v16h bz1 = ld_frag(&sBz[cur][(waveN * 32 + 16) * LDS_STRIDE], lane);
        v16h bh0 = ld_frag(&sBh[cur][(waveN * 32 +  0) * LDS_STRIDE], lane);
        v16h bh1 = ld_frag(&sBh[cur][(waveN * 32 + 16) * LDS_STRIDE], lane);

        ak[0][0] = wmma_f16(a0, bz0, ak[0][0]);
        ak[0][1] = wmma_f16(a0, bz1, ak[0][1]);
        ak[1][0] = wmma_f16(a1, bz0, ak[1][0]);
        ak[1][1] = wmma_f16(a1, bz1, ak[1][1]);
        ap[0][0] = wmma_f16(a0, bh0, ap[0][0]);
        ap[0][1] = wmma_f16(a0, bh1, ap[0][1]);
        ap[1][0] = wmma_f16(a1, bh0, ap[1][0]);
        ap[1][1] = wmma_f16(a1, bh1, ap[1][1]);
        __syncthreads();              // reads done before buf[cur] refilled (s+2)
    }

    // ---- epilogue: C/D layout -> row = v + (lane>=16 ? 8:0), col = lane&15 ----
    const int rowOff = (lane & 16) ? 8 : 0;
#pragma unroll
    for (int i = 0; i < 2; ++i) {
#pragma unroll
        for (int j = 0; j < 2; ++j) {
            const int rBase = mBase + waveM * 32 + i * 16;
            const int nCol  = nBase + waveN * 32 + j * 16 + (lane & 15);
            const float bzv = bz[nCol];
            const float bhv = bh[nCol];
#pragma unroll
            for (int v = 0; v < 8; ++v) {
                int   m  = rBase + rowOff + v;
                float kv = ak[i][j][v] + bzv;
                float pv = ap[i][j][v] + bhv;
                float cv = -sp(kv);
                float vv = -sp(-kv) + log_g(pv);
                int b = m >> 12;                       // T_ = 4096
                int t = m & (T_ - 1);
                cbuf[(size_t)m * H_ + nCol] = cv;
                out[((size_t)b * (T_ + 1) + t + 1) * H_ + nCol] = vv;
            }
        }
    }
}

// ---------- Kernel 2: out[b,0,h] = g(h0) ----------
__global__ void MinGRU_h0_kernel(const float* __restrict__ h0, float* __restrict__ out) {
    int gid = blockIdx.x * blockDim.x + threadIdx.x;  // B*H threads
    int h = gid & (H_ - 1);
    int b = gid >> 10;
    out[(size_t)b * (T_ + 1) * H_ + h] = g_fun(h0[b * H_ + h]);
}

// ---------- Kernel 3: per-chunk local scan -> (C, V) summaries ----------
__global__ void MinGRU_scan1_kernel(const float* __restrict__ cbuf,
                                    const float* __restrict__ out,
                                    float* __restrict__ Carr, float* __restrict__ Varr) {
    int gid = blockIdx.x * blockDim.x + threadIdx.x;  // B*NC*H
    int h = gid & (H_ - 1);
    int tmp = gid >> 10;
    int q = tmp & (NC - 1);
    int b = tmp >> 5;
    float C = 0.0f, V = -__builtin_inff();
    int t0 = q * TC;
    for (int t = t0; t < t0 + TC; ++t) {
        float c = cbuf[((size_t)(b << 12) + t) * H_ + h];
        float v = out[((size_t)b * (T_ + 1) + t + 1) * H_ + h];
        C += c;
        V = lae(V + c, v);
    }
    Carr[gid] = C;
    Varr[gid] = V;
}

// ---------- Kernel 4: inter-chunk scan -> L at chunk starts ----------
__global__ void MinGRU_scan2_kernel(const float* __restrict__ h0,
                                    const float* __restrict__ Carr,
                                    const float* __restrict__ Varr,
                                    float* __restrict__ Lst) {
    int gid = blockIdx.x * blockDim.x + threadIdx.x;  // B*H
    int h = gid & (H_ - 1);
    int b = gid >> 10;
    float L = log_g(h0[b * H_ + h]);
    for (int q = 0; q < NC; ++q) {
        int idx = ((b * NC + q) << 10) + h;
        Lst[idx] = L;
        L = lae(Carr[idx] + L, Varr[idx]);
    }
}

// ---------- Kernel 5: fixup pass, write exp(L) over the v values ----------
__global__ void MinGRU_scan3_kernel(const float* __restrict__ cbuf,
                                    float* __restrict__ out,
                                    const float* __restrict__ Lst) {
    int gid = blockIdx.x * blockDim.x + threadIdx.x;  // B*NC*H
    int h = gid & (H_ - 1);
    int tmp = gid >> 10;
    int q = tmp & (NC - 1);
    int b = tmp >> 5;
    float L = Lst[gid];
    int t0 = q * TC;
    for (int t = t0; t < t0 + TC; ++t) {
        float c = cbuf[((size_t)(b << 12) + t) * H_ + h];
        size_t oidx = ((size_t)b * (T_ + 1) + t + 1) * H_ + h;
        float v = out[oidx];
        L = lae(c + L, v);
        out[oidx] = expf(L);
    }
}

extern "C" void kernel_launch(void* const* d_in, const int* in_sizes, int n_in,
                              void* d_out, int out_size, void* d_ws, size_t ws_size,
                              hipStream_t stream) {
    const float* x  = (const float*)d_in[0];
    const float* h0 = (const float*)d_in[1];
    const float* Wz = (const float*)d_in[2];
    const float* bz = (const float*)d_in[3];
    const float* Wh = (const float*)d_in[4];
    const float* bh = (const float*)d_in[5];
    float* out = (float*)d_out;

    float* cbuf = (float*)d_ws;                               // B*T*H   (128 MB)
    float* Carr = cbuf + (size_t)B_ * T_ * H_;                // B*NC*H  (1 MB)
    float* Varr = Carr + (size_t)B_ * NC * H_;
    float* Lst  = Varr + (size_t)B_ * NC * H_;
    _Float16* xh  = (_Float16*)(Lst + (size_t)B_ * NC * H_);  // B*T*D f16 (64 MB)
    _Float16* wzt = xh + (size_t)B_ * T_ * D_;                // [H,D] f16 (2 MB)
    _Float16* wht = wzt + (size_t)D_ * H_;

    MinGRU_cvtx_kernel<<<(B_ * T_ * D_) / (256 * 8), 256, 0, stream>>>(x, xh);
    MinGRU_cvtw_kernel<<<dim3(D_ / 32, H_ / 32, 2), 256, 0, stream>>>(Wz, Wh, wzt, wht);

    dim3 grid(H_ / 64, (B_ * T_) / 128);                      // 16 x 256 blocks
    MinGRU_gemm_kernel<<<grid, 256, 0, stream>>>(xh, wzt, wht, bz, bh, cbuf, out);
    MinGRU_h0_kernel  <<<(B_ * H_) / 256, 256, 0, stream>>>(h0, out);
    MinGRU_scan1_kernel<<<(B_ * NC * H_) / 256, 256, 0, stream>>>(cbuf, out, Carr, Varr);
    MinGRU_scan2_kernel<<<(B_ * H_) / 256, 256, 0, stream>>>(h0, Carr, Varr, Lst);
    MinGRU_scan3_kernel<<<(B_ * NC * H_) / 256, 256, 0, stream>>>(cbuf, out, Lst);
}
